// TransformerLayer_31782757990579
// MI455X (gfx1250) — matrix operand order repaired and checked by reading, not compiled, not run on hardware
//
#include <hip/hip_runtime.h>
#include <math.h>

// ---------------------------------------------------------------------------
// Types for CDNA5 WMMA (wave32, 16x16x32 bf16 -> f32)
// ---------------------------------------------------------------------------
typedef __attribute__((ext_vector_type(16))) __bf16 v16bf;
typedef __attribute__((ext_vector_type(8)))  __bf16 v8bf;
typedef __attribute__((ext_vector_type(8)))  float  v8f;

#if defined(__HIP_DEVICE_COMPILE__) && defined(__gfx1250__)
#define ASYNC_PATH 1
#else
#define ASYNC_PATH 0
#endif

__device__ __forceinline__ __bf16 f2bf(float f) {
  unsigned u = __float_as_uint(f);
  unsigned short s = (unsigned short)((u + 0x7FFFu + ((u >> 16) & 1u)) >> 16);
  return __builtin_bit_cast(__bf16, s);
}
__device__ __forceinline__ float bf2f(__bf16 v) {
  unsigned short s = __builtin_bit_cast(unsigned short, v);
  return __uint_as_float(((unsigned)s) << 16);
}

__device__ __forceinline__ v8f wmma_bf16(v16bf a, v16bf b, v8f c) {
  return __builtin_amdgcn_wmma_f32_16x16x32_bf16(false, a, false, b, (short)0, c,
                                                 false, false);
}

// A-fragment (16x32, MxK): lane L<16 holds row M=L, K in {0..7, 16..23};
// lanes 16..31 hold K in {8..15, 24..31}.  base = row start (+k0).
__device__ __forceinline__ v16bf frag_a(const __bf16* base, int lhalf) {
  v8bf lo = *(const v8bf*)(base + 8 * lhalf);
  v8bf hi = *(const v8bf*)(base + 8 * lhalf + 16);
  return __builtin_shufflevector(lo, hi, 0, 1, 2, 3, 4, 5, 6, 7,
                                 8, 9, 10, 11, 12, 13, 14, 15);
}

// B-fragment (32x16, KxN): lane L holds column N=L&15, K in {0..7, 16..23};
// lanes 16..31 hold K in {8..15, 24..31}.  base = column-source row start (+k0).
__device__ __forceinline__ v16bf frag_b(const __bf16* base, int lhalf) {
  v8bf lo = *(const v8bf*)(base + 8 * lhalf);
  v8bf hi = *(const v8bf*)(base + 8 * lhalf + 16);
  return __builtin_shufflevector(lo, hi, 0, 1, 2, 3, 4, 5, 6, 7,
                                 8, 9, 10, 11, 12, 13, 14, 15);
}

// 16-byte global -> LDS async copy (CDNA5 GLOBAL_LOAD_ASYNC_TO_LDS_B128).
// vDst = wave-relative LDS byte offset (low 32 bits of generic LDS pointer),
// vaddr = 64-bit global address.  Tracked by ASYNCcnt.
__device__ __forceinline__ void async_copy16(const __bf16* g, __bf16* l) {
#if ASYNC_PATH
  unsigned lofs = (unsigned)(unsigned long long)l;
  asm volatile("global_load_async_to_lds_b128 %0, %1, off"
               :
               : "v"(lofs), "v"(g)
               : "memory");
#else
  *(v8bf*)l = *(const v8bf*)g;
#endif
}

__device__ __forceinline__ void wait_async0() {
#if ASYNC_PATH
  asm volatile("s_wait_asynccnt 0" ::: "memory");
#endif
}

// ---------------------------------------------------------------------------
// 1) Fold LoRA into weights:  W_eff = W + (1/R) * B @ A   (bf16 output)
// ---------------------------------------------------------------------------
__global__ void k_merge_w(const float* __restrict__ W, const float* __restrict__ A,
                          const float* __restrict__ Bm, __bf16* __restrict__ out,
                          int OUT, int IN) {
  int idx = blockIdx.x * blockDim.x + threadIdx.x;
  if (idx >= OUT * IN) return;
  int o = idx / IN, i = idx - o * IN;
  float w = W[idx];
  if (A != nullptr) {
    float acc = 0.f;
#pragma unroll
    for (int r = 0; r < 8; ++r) acc += Bm[o * 8 + r] * A[r * IN + i];
    w += acc * 0.125f;  // SCALING = 1/8
  }
  out[idx] = f2bf(w);
}

// ---------------------------------------------------------------------------
// 2) T5 relative position bias -> d_out second region, layout [H, T, T]
// ---------------------------------------------------------------------------
__global__ void k_posbias(const float* __restrict__ rel_bias, float* __restrict__ pb,
                          int T) {
  int q = blockIdx.y;
  int kk = blockIdx.x * blockDim.x + threadIdx.x;
  if (kk >= T) return;
  int rel = kk - q;  // key_pos - query_pos
  const int nb = 16, max_exact = 8;
  int bucket = (rel > 0) ? nb : 0;
  int a = rel < 0 ? -rel : rel;
  int val;
  if (a < max_exact) {
    val = a;
  } else {
    float lf = logf((float)a / (float)max_exact) / logf(128.f / 8.f) *
               (float)(nb - max_exact);
    int large = max_exact + (int)lf;
    val = large < nb - 1 ? large : nb - 1;
  }
  bucket += val;
#pragma unroll
  for (int h = 0; h < 16; ++h)
    pb[(size_t)h * T * T + (size_t)q * T + kk] = rel_bias[bucket * 16 + h];
}

// ---------------------------------------------------------------------------
// 3) RMSNorm (fp32 in, bf16 out), one block (256 thr) per row of D=1024
// ---------------------------------------------------------------------------
__global__ __launch_bounds__(256) void k_rmsnorm(const float* __restrict__ x,
                                                 const float* __restrict__ w,
                                                 __bf16* __restrict__ y, int D) {
  int row = blockIdx.x;
  const float* xr = x + (size_t)row * D;
  float s = 0.f;
  for (int i = threadIdx.x; i < D; i += 256) {
    float v = xr[i];
    s += v * v;
  }
#pragma unroll
  for (int off = 16; off > 0; off >>= 1) s += __shfl_xor(s, off, 32);
  __shared__ float red[8];
  if ((threadIdx.x & 31) == 0) red[threadIdx.x >> 5] = s;
  __syncthreads();
  float t = 0.f;
#pragma unroll
  for (int i = 0; i < 8; ++i) t += red[i];
  float r = rsqrtf(t / (float)D + 1e-6f);
  for (int i = threadIdx.x; i < D; i += 256)
    y[(size_t)row * D + i] = f2bf(xr[i] * r * w[i]);
}

// ---------------------------------------------------------------------------
// 4) bf16 WMMA GEMM:  C[M,N] = A[M,K] * W[N,K]^T
//    256 threads = 8 waves; 128x128 block tile, BK=32; wave tile 32x64.
//    Double-buffered LDS with CDNA5 async global->LDS copies (ASYNCcnt).
//    MODE 0: write bf16;  MODE 1: outF = resid + C (fp32).
// ---------------------------------------------------------------------------
#define LDSW 40  // LDS row stride (elems) = 80B: 16B-aligned, conflict-padded

template <int MODE, int N, int K>
__global__ __launch_bounds__(256) void k_gemm_bf16(
    const __bf16* __restrict__ Amat, const __bf16* __restrict__ Wmat, int M,
    __bf16* __restrict__ outB, float* __restrict__ outF,
    const float* __restrict__ resid) {
#if ASYNC_PATH
  __shared__ __bf16 As[2][128 * LDSW];
  __shared__ __bf16 Bs[2][128 * LDSW];
#else
  __shared__ __bf16 As[1][128 * LDSW];
  __shared__ __bf16 Bs[1][128 * LDSW];
#endif
  int tid = threadIdx.x;
  int lane = tid & 31, wave = tid >> 5;
  int wm = wave >> 1, wn = wave & 1;
  int lhalf = lane >> 4, lmod = lane & 15;
  int m0 = blockIdx.y * 128, n0 = blockIdx.x * 128;

  // tile copy assignment: each thread moves 16 elems (2 x b128) of A and of B
  int lr = tid >> 1;          // 0..127: tile row
  int ls = (tid & 1) * 16;    // 0 or 16: 16-element half-row
  const __bf16* gA = Amat + (size_t)(m0 + lr) * K + ls;
  const __bf16* gB = Wmat + (size_t)(n0 + lr) * K + ls;

  v8f zero = {};
  v8f acc[2][4];
#pragma unroll
  for (int i = 0; i < 2; ++i)
#pragma unroll
    for (int j = 0; j < 4; ++j) acc[i][j] = zero;

  const int NK = K / 32;
#if ASYNC_PATH
  async_copy16(gA, &As[0][lr * LDSW + ls]);
  async_copy16(gA + 8, &As[0][lr * LDSW + ls + 8]);
  async_copy16(gB, &Bs[0][lr * LDSW + ls]);
  async_copy16(gB + 8, &Bs[0][lr * LDSW + ls + 8]);
#endif

  for (int kt = 0; kt < NK; ++kt) {
#if ASYNC_PATH
    int cur = kt & 1;
    wait_async0();
    __syncthreads();
    if (kt + 1 < NK) {
      int nxt = (kt + 1) & 1;
      const __bf16* ga = gA + (kt + 1) * 32;
      const __bf16* gb = gB + (kt + 1) * 32;
      async_copy16(ga, &As[nxt][lr * LDSW + ls]);
      async_copy16(ga + 8, &As[nxt][lr * LDSW + ls + 8]);
      async_copy16(gb, &Bs[nxt][lr * LDSW + ls]);
      async_copy16(gb + 8, &Bs[nxt][lr * LDSW + ls + 8]);
    }
#else
    const int cur = 0;
    __syncthreads();
    *(v8bf*)(&As[0][lr * LDSW + ls]) = *(const v8bf*)(gA + kt * 32);
    *(v8bf*)(&As[0][lr * LDSW + ls + 8]) = *(const v8bf*)(gA + kt * 32 + 8);
    *(v8bf*)(&Bs[0][lr * LDSW + ls]) = *(const v8bf*)(gB + kt * 32);
    *(v8bf*)(&Bs[0][lr * LDSW + ls + 8]) = *(const v8bf*)(gB + kt * 32 + 8);
    __syncthreads();
#endif
    v16bf af[2], bfg[4];
#pragma unroll
    for (int i = 0; i < 2; ++i)
      af[i] = frag_a(&As[cur][(wm * 32 + i * 16 + lmod) * LDSW], lhalf);
#pragma unroll
    for (int j = 0; j < 4; ++j)
      bfg[j] = frag_b(&Bs[cur][(wn * 64 + j * 16 + lmod) * LDSW], lhalf);
#pragma unroll
    for (int i = 0; i < 2; ++i)
#pragma unroll
      for (int j = 0; j < 4; ++j) acc[i][j] = wmma_bf16(af[i], bfg[j], acc[i][j]);
  }

#pragma unroll
  for (int i = 0; i < 2; ++i)
#pragma unroll
    for (int j = 0; j < 4; ++j) {
      int mbase = m0 + wm * 32 + i * 16 + 8 * lhalf;
      int nn = n0 + wn * 64 + j * 16 + lmod;
#pragma unroll
      for (int r = 0; r < 8; ++r) {
        size_t idx = (size_t)(mbase + r) * N + nn;
        if (MODE == 0)
          outB[idx] = f2bf(acc[i][j][r]);
        else
          outF[idx] = resid[idx] + acc[i][j][r];
      }
    }
}

// ---------------------------------------------------------------------------
// 5) V transpose: v[B,T,H,64] -> vt[B,H,64,T]  (bf16)
// ---------------------------------------------------------------------------
__global__ void k_transpose_v(const __bf16* __restrict__ v, __bf16* __restrict__ vt,
                              int Bb, int T) {
  size_t idx = (size_t)blockIdx.x * blockDim.x + threadIdx.x;
  size_t total = (size_t)Bb * T * 1024;
  if (idx >= total) return;
  size_t b = idx / ((size_t)T * 1024);
  size_t rem = idx - b * (size_t)T * 1024;
  int t = (int)(rem / 1024);
  int c = (int)(rem & 1023);
  int h = c >> 6, d = c & 63;
  vt[((b * 16 + h) * 64 + d) * (size_t)T + t] = v[idx];
}

// ---------------------------------------------------------------------------
// 6) Flash attention, bf16 WMMA, online softmax.
//    grid: (T/64, B*H); block 128 = 4 waves; each wave: 16 query rows.
//    S = Q K^T * 0.125 + pos_bias; mask; P*V with V^T layout.
//    P staging LDS is wave-private: LDS ops are in-order per wave (DScnt),
//    so no block barrier is needed -- only a compiler reordering fence.
// ---------------------------------------------------------------------------
__global__ __launch_bounds__(128) void k_flash_attn(
    const __bf16* __restrict__ q, const __bf16* __restrict__ k,
    const __bf16* __restrict__ vt, const float* __restrict__ posb,
    const int* __restrict__ mask, __bf16* __restrict__ o, int T) {
  int bh = blockIdx.y;
  int b = bh >> 4, h = bh & 15;
  int tid = threadIdx.x, lane = tid & 31, wave = tid >> 5;
  int lhalf = lane >> 4, lmod = lane & 15;
  int q0 = blockIdx.x * 64 + wave * 16;

  const __bf16* qp = q + (size_t)b * T * 1024 + h * 64;
  const __bf16* kp = k + (size_t)b * T * 1024 + h * 64;
  const __bf16* vtp = vt + (size_t)(b * 16 + h) * 64 * (size_t)T;
  const float* pb = posb + (size_t)h * T * T;
  const int* mk = mask + (size_t)b * T * T;

  __shared__ __bf16 plds[4][16 * LDSW];
  __bf16* pw = plds[wave];

  v16bf aq0 = frag_a(qp + (size_t)(q0 + lmod) * 1024, lhalf);       // dh 0..31
  v16bf aq1 = frag_a(qp + (size_t)(q0 + lmod) * 1024 + 32, lhalf);  // dh 32..63

  // per-row bias/mask bases for the 8 rows this lane touches
  const float* pbrow[8];
  const int* mkrow[8];
#pragma unroll
  for (int r = 0; r < 8; ++r) {
    int qq = q0 + r + 8 * lhalf;
    pbrow[r] = pb + (size_t)qq * T;
    mkrow[r] = mk + (size_t)qq * T;
  }

  float mstat[8], lstat[8];
  v8f zero = {};
  v8f oacc[4];
#pragma unroll
  for (int r = 0; r < 8; ++r) {
    mstat[r] = -1e30f;
    lstat[r] = 0.f;
  }
#pragma unroll
  for (int j = 0; j < 4; ++j) oacc[j] = zero;

  for (int j = 0; j < T; j += 32) {
    // ---- S = Q K^T over 32 keys (two 16-key n-tiles, two dh k-steps) ----
    v8f s0 = zero, s1 = zero;
    {
      const __bf16* kb0 = kp + (size_t)(j + lmod) * 1024;
      const __bf16* kb1 = kp + (size_t)(j + 16 + lmod) * 1024;
      s0 = wmma_bf16(aq0, frag_b(kb0, lhalf), s0);
      s0 = wmma_bf16(aq1, frag_b(kb0 + 32, lhalf), s0);
      s1 = wmma_bf16(aq0, frag_b(kb1, lhalf), s1);
      s1 = wmma_bf16(aq1, frag_b(kb1 + 32, lhalf), s1);
    }
    // ---- scale, bias, mask ----
    float sv0[8], sv1[8];
#pragma unroll
    for (int r = 0; r < 8; ++r) {
      int c0 = j + lmod, c1 = j + 16 + lmod;
      float x0 = s0[r] * 0.125f + pbrow[r][c0];
      float x1 = s1[r] * 0.125f + pbrow[r][c1];
      if (mkrow[r][c0] == 0) x0 = -1e9f;
      if (mkrow[r][c1] == 0) x1 = -1e9f;
      sv0[r] = x0;
      sv1[r] = x1;
    }
    // ---- online softmax (row stats replicated across each 16-lane half) ----
    float pn0[8], pn1[8], alpha[8];
#pragma unroll
    for (int r = 0; r < 8; ++r) {
      float mx = fmaxf(sv0[r], sv1[r]);
#pragma unroll
      for (int off = 8; off > 0; off >>= 1) mx = fmaxf(mx, __shfl_xor(mx, off, 16));
      float mnew = fmaxf(mstat[r], mx);
      float a = __expf(mstat[r] - mnew);
      float p0 = __expf(sv0[r] - mnew);
      float p1 = __expf(sv1[r] - mnew);
      float rs = p0 + p1;
#pragma unroll
      for (int off = 8; off > 0; off >>= 1) rs += __shfl_xor(rs, off, 16);
      lstat[r] = lstat[r] * a + rs;
      mstat[r] = mnew;
      alpha[r] = a;
      pn0[r] = p0;
      pn1[r] = p1;
    }
#pragma unroll
    for (int n = 0; n < 4; ++n) {
      v8f t = oacc[n];
#pragma unroll
      for (int r = 0; r < 8; ++r) t[r] *= alpha[r];
      oacc[n] = t;
    }
    // ---- P (C-layout) -> wave-private LDS -> A-layout fragment ----
#pragma unroll
    for (int r = 0; r < 8; ++r) {
      pw[(r + 8 * lhalf) * LDSW + lmod] = f2bf(pn0[r]);
      pw[(r + 8 * lhalf) * LDSW + 16 + lmod] = f2bf(pn1[r]);
    }
    __builtin_amdgcn_wave_barrier();       // keep compiler from reordering
    asm volatile("" ::: "memory");         // (LDS is in-order per wave in HW)
    v16bf ap = frag_a(pw + lmod * LDSW, lhalf);
    // ---- O += P * V  (B-fragments contiguous from V^T) ----
#pragma unroll
    for (int n = 0; n < 4; ++n) {
      v16bf bv = frag_b(vtp + (size_t)(n * 16 + lmod) * T + j, lhalf);
      oacc[n] = wmma_bf16(ap, bv, oacc[n]);
    }
  }
  // ---- finalize: O /= l, write o[B,T,1024] bf16 ----
#pragma unroll
  for (int n = 0; n < 4; ++n)
#pragma unroll
    for (int r = 0; r < 8; ++r) {
      int qq = q0 + r + 8 * lhalf;
      o[((size_t)b * T + qq) * 1024 + h * 64 + n * 16 + lmod] =
          f2bf(oacc[n][r] / lstat[r]);
    }
}

// ---------------------------------------------------------------------------
// 7) GEGLU: g = p1 * gelu_tanh(p2);  h[M,4096] -> g[M,2048]
// ---------------------------------------------------------------------------
__global__ void k_geglu(const __bf16* __restrict__ hb, __bf16* __restrict__ g,
                        int M) {
  size_t idx = (size_t)blockIdx.x * blockDim.x + threadIdx.x;
  size_t total = (size_t)M * 2048;
  if (idx >= total) return;
  size_t row = idx / 2048;
  int c = (int)(idx & 2047);
  float p1 = bf2f(hb[row * 4096 + c]);
  float p2 = bf2f(hb[row * 4096 + 2048 + c]);
  float inner = 0.7978845608028654f * (p2 + 0.044715f * p2 * p2 * p2);
  float gl = 0.5f * p2 * (1.f + tanhf(inner));
  g[idx] = f2bf(p1 * gl);
}

// ---------------------------------------------------------------------------
// Host orchestration
// ---------------------------------------------------------------------------
extern "C" void kernel_launch(void* const* d_in, const int* in_sizes, int n_in,
                              void* d_out, int out_size, void* d_ws, size_t ws_size,
                              hipStream_t stream) {
  const int Bb = 2, T = 2048, D = 1024, M = Bb * T;

  const float* x = (const float*)d_in[0];
  const int* xmask = (const int*)d_in[1];
  const float* n1w = (const float*)d_in[3];
  const float* n3w = (const float*)d_in[4];
  const float* wqW = (const float*)d_in[5];
  const float* wqA = (const float*)d_in[6];
  const float* wqB = (const float*)d_in[7];
  const float* wkW = (const float*)d_in[8];
  const float* wvW = (const float*)d_in[9];
  const float* wvA = (const float*)d_in[10];
  const float* wvB = (const float*)d_in[11];
  const float* fcW = (const float*)d_in[12];
  const float* fcA = (const float*)d_in[13];
  const float* fcB = (const float*)d_in[14];
  const float* relb = (const float*)d_in[15];
  const float* w1W = (const float*)d_in[16];
  const float* w1A = (const float*)d_in[17];
  const float* w1B = (const float*)d_in[18];
  const float* w2W = (const float*)d_in[19];
  const float* w2A = (const float*)d_in[20];
  const float* w2B = (const float*)d_in[21];

  float* out_x = (float*)d_out;
  float* out_pb = out_x + (size_t)Bb * T * D;  // [H,1,T,T]

  char* ws = (char*)d_ws;
  size_t off = 0;
  auto take = [&](size_t bytes) -> char* {
    char* p = ws + off;
    off = (off + bytes + 255) & ~(size_t)255;
    return p;
  };
  __bf16* wqE = (__bf16*)take((size_t)D * D * 2);
  __bf16* wkE = (__bf16*)take((size_t)D * D * 2);
  __bf16* wvE = (__bf16*)take((size_t)D * D * 2);
  __bf16* fcE = (__bf16*)take((size_t)D * D * 2);
  __bf16* w1E = (__bf16*)take((size_t)4 * D * D * 2);
  __bf16* w2E = (__bf16*)take((size_t)D * 2 * D * 2);
  __bf16* y1 = (__bf16*)take((size_t)M * D * 2);
  __bf16* qb = (__bf16*)take((size_t)M * D * 2);
  __bf16* kb = (__bf16*)take((size_t)M * D * 2);
  __bf16* vb = (__bf16*)take((size_t)M * D * 2);
  __bf16* vtb = (__bf16*)take((size_t)M * D * 2);
  __bf16* ob = (__bf16*)take((size_t)M * D * 2);
  float* x2 = (float*)take((size_t)M * D * 4);
  __bf16* y3 = (__bf16*)take((size_t)M * D * 2);
  __bf16* hb = (__bf16*)take((size_t)M * 4 * D * 2);
  __bf16* gb = (__bf16*)take((size_t)M * 2 * D * 2);

  // 1) fold LoRA into weights (bf16)
  {
    int thr = 256;
    k_merge_w<<<(D * D + thr - 1) / thr, thr, 0, stream>>>(wqW, wqA, wqB, wqE, D, D);
    k_merge_w<<<(D * D + thr - 1) / thr, thr, 0, stream>>>(wkW, nullptr, nullptr,
                                                           wkE, D, D);
    k_merge_w<<<(D * D + thr - 1) / thr, thr, 0, stream>>>(wvW, wvA, wvB, wvE, D, D);
    k_merge_w<<<(D * D + thr - 1) / thr, thr, 0, stream>>>(fcW, fcA, fcB, fcE, D, D);
    k_merge_w<<<(4 * D * D + thr - 1) / thr, thr, 0, stream>>>(w1W, w1A, w1B, w1E,
                                                               4 * D, D);
    k_merge_w<<<(2 * D * D + thr - 1) / thr, thr, 0, stream>>>(w2W, w2A, w2B, w2E,
                                                               D, 2 * D);
  }
  // 2) relative position bias -> d_out (read later by attention)
  k_posbias<<<dim3(T / 256, T), 256, 0, stream>>>(relb, out_pb, T);
  // 3) y1 = rmsnorm(x) (bf16)
  k_rmsnorm<<<M, 256, 0, stream>>>(x, n1w, y1, D);
  // 4) Q, K, V projections (bf16 WMMA GEMMs, compile-time N/K)
  dim3 gD(D / 128, M / 128);
  k_gemm_bf16<0, 1024, 1024><<<gD, 256, 0, stream>>>(y1, wqE, M, qb, nullptr,
                                                     nullptr);
  k_gemm_bf16<0, 1024, 1024><<<gD, 256, 0, stream>>>(y1, wkE, M, kb, nullptr,
                                                     nullptr);
  k_gemm_bf16<0, 1024, 1024><<<gD, 256, 0, stream>>>(y1, wvE, M, vb, nullptr,
                                                     nullptr);
  // 5) V^T for contiguous B-fragments in P*V
  k_transpose_v<<<(int)(((size_t)M * D + 255) / 256), 256, 0, stream>>>(vb, vtb, Bb,
                                                                        T);
  // 6) flash attention
  k_flash_attn<<<dim3(T / 64, Bb * 16), 128, 0, stream>>>(qb, kb, vtb, out_pb,
                                                          xmask, ob, T);
  // 7) x2 = x + o @ fc^T  (fp32)
  k_gemm_bf16<1, 1024, 1024><<<gD, 256, 0, stream>>>(ob, fcE, M, nullptr, x2, x);
  // 8) y3 = rmsnorm(x2)
  k_rmsnorm<<<M, 256, 0, stream>>>(x2, n3w, y3, D);
  // 9) h = y3 @ w1^T  [M, 4096]
  k_gemm_bf16<0, 4096, 1024><<<dim3(4 * D / 128, M / 128), 256, 0, stream>>>(
      y3, w1E, M, hb, nullptr, nullptr);
  // 10) g = p1 * gelu(p2)  [M, 2048]
  k_geglu<<<(int)(((size_t)M * 2 * D + 255) / 256), 256, 0, stream>>>(hb, gb, M);
  // 11) out = x2 + g @ w2^T  (fp32 -> d_out)
  k_gemm_bf16<1, 1024, 2048><<<dim3(D / 128, M / 128), 256, 0, stream>>>(
      gb, w2E, M, nullptr, out_x, x2);
}
